// MultiheadAttention_46755013984982
// MI455X (gfx1250) — compile-verified, matrix operand order checked by reading, not run on hardware
//
#include <hip/hip_runtime.h>
#include <hip/hip_bf16.h>

typedef __attribute__((ext_vector_type(16))) _Float16 v16h;
typedef __attribute__((ext_vector_type(8)))  _Float16 v8h;
typedef __attribute__((ext_vector_type(8)))  float    v8f;

#define N_SEQ 1024

// ---------------------------------------------------------------------------
// Generic [rows,64] @ [64,64]^T GEMM via WMMA f16 (f32 accumulate).
// O[r][c] = sum_k X[r][k] * W[c][k]  (+ bias[c] if bias != nullptr)
// One 16x16 output tile per wave; K=64 -> 2 chunks of 32.
// ---------------------------------------------------------------------------
__global__ __launch_bounds__(256) void gemm64_wmma(
    const float* __restrict__ X, const float* __restrict__ W,
    const float* __restrict__ bias, float* __restrict__ O, int rows)
{
    const int lane = threadIdx.x & 31;
    const int wave = threadIdx.x >> 5;
    const int tile = blockIdx.x * 8 + wave;
    const int rowTiles = rows >> 4;
    if (tile >= rowTiles * 4) return;           // uniform per wave
    const int r0 = (tile >> 2) << 4;
    const int c0 = (tile & 3) << 4;

    const int row = lane & 15;                  // A row / B col / D col index
    const int hi  = lane >> 4;                  // lane half select
    const int kb  = hi * 8;

    v8f acc = {};
#pragma unroll
    for (int kc = 0; kc < 2; ++kc) {
        const int k0 = kc * 32;
        const float* xp = X + (size_t)(r0 + row) * 64 + k0 + kb;
        const float* wp = W + (size_t)(c0 + row) * 64 + k0 + kb;
        v16h a, bf;
#pragma unroll
        for (int e = 0; e < 8; ++e) {
            // 16-bit A 16x32 layout: lane half kb, VGPR0-3 -> K=kb..kb+7,
            // VGPR4-7 -> K=16+kb..16+kb+7 (elements 8..15).
            a[e]    = (_Float16)xp[e];
            a[e+8]  = (_Float16)xp[e + 16];
            bf[e]   = (_Float16)wp[e];
            bf[e+8] = (_Float16)wp[e + 16];
        }
        acc = __builtin_amdgcn_wmma_f32_16x16x32_f16(
            false, a, false, bf, (short)0, acc, false, false);
    }
    const float bv = bias ? bias[c0 + row] : 0.0f;   // hoisted (c invariant in j)
#pragma unroll
    for (int j = 0; j < 8; ++j) {
        const int r = r0 + j + hi * 8;          // D: VGPR j holds M=j (+8 for hi lanes)
        O[(size_t)r * 64 + c0 + row] = acc[j] + bv;
    }
}

// ---------------------------------------------------------------------------
// Fused: relative-position scores + folded BN/ReLU MLP + softmax + attn @ V.
// One block per (b, h, 16-row n-tile): 2*4*64 = 512 blocks of 256 threads.
// Streams pos exactly once (537 MB -> HBM-bound floor ~23 us @ 23.3 TB/s).
// LDS ~133 KB -> 2 workgroups/WGP; kh is read through the 192 MB L2 instead
// (512 KB working set) to keep occupancy up for latency hiding.
// ---------------------------------------------------------------------------
__global__ __launch_bounds__(256) void attn_fused(
    const float* __restrict__ qh, const float* __restrict__ kh,
    const float* __restrict__ vh, const float* __restrict__ pos,
    const float* __restrict__ w1, const float* __restrict__ b1,
    const float* __restrict__ bn1_g, const float* __restrict__ bn1_b,
    const float* __restrict__ bn1_m, const float* __restrict__ bn1_v,
    const float* __restrict__ w2, const float* __restrict__ b2,
    const float* __restrict__ bn2_g, const float* __restrict__ bn2_b,
    const float* __restrict__ bn2_m, const float* __restrict__ bn2_v,
    float* __restrict__ y)
{
    alignas(16) __shared__ float    s_s [16 * N_SEQ];   // 64 KB scores s_s[n*1024+m]
    alignas(16) __shared__ _Float16 vhT [16 * N_SEQ];   // 32 KB vhT[d*1024+m] (B frags)
    alignas(16) __shared__ _Float16 attn_s[16 * N_SEQ]; // 32 KB attn_s[n*1024+m] (A frags)
    alignas(16) __shared__ float    qh_s[16 * 16];
    alignas(16) __shared__ float    ca[16], cb[16], w2c[16];
    __shared__ float sbias;
    __shared__ float y_tile[16 * 16];

    const int t  = threadIdx.x;
    const int bt = blockIdx.x;
    const int b  = bt >> 8;
    const int h  = (bt >> 6) & 3;
    const int n0 = (bt & 63) << 4;

    // Fold attn_mlp constants: z = relu(x*ca + cb); s = dot(z, w2c) + sbias
    if (t < 16) {
        const float g1 = bn1_g[t] * rsqrtf(bn1_v[t] + 1e-5f);
        ca[t]  = w1[t] * g1;
        cb[t]  = (b1[t] - bn1_m[t]) * g1 + bn1_b[t];
        const float g2 = bn2_g[0] * rsqrtf(bn2_v[0] + 1e-5f);
        w2c[t] = w2[t] * g2;
        if (t == 0) sbias = (b2[0] - bn2_m[0]) * g2 + bn2_b[0];
    }
    qh_s[t]   = qh[(size_t)(b * N_SEQ + n0 + (t >> 4)) * 64 + h * 16 + (t & 15)];
    y_tile[t] = 0.0f;
    for (int i = t; i < N_SEQ * 16; i += 256) {
        const int m = i >> 4, d = i & 15;
        vhT[d * N_SEQ + m] =
            (_Float16)vh[(size_t)(b * N_SEQ + m) * 64 + h * 16 + d];
    }
    __syncthreads();

    // ---- score phase: quad (4 lanes) per (n,m) pair, float4 over d ----
    const int lane = t & 31, wave = t >> 5;
    const int quad = lane >> 2, lq = lane & 3, d0 = lq << 2;
    const float4 cav = *(const float4*)&ca [d0];
    const float4 cbv = *(const float4*)&cb [d0];
    const float4 w2v = *(const float4*)&w2c[d0];
    const float* posP = pos + ((size_t)(b * N_SEQ + n0) * N_SEQ) * 64 + h * 16 + d0;
    const float* khP  = kh  + (size_t)b * N_SEQ * 64 + h * 16 + d0;

#pragma unroll 8
    for (int it = 0; it < 256; ++it) {
        const int pair = wave * 8 + it * 64 + quad;   // pair = n*1024 + m
        const int n = pair >> 10, m = pair & (N_SEQ - 1);
        const float4 pv = *(const float4*)(posP + (size_t)pair * 64);
        const float4 kv = *(const float4*)(khP + (size_t)m * 64);   // L2-resident
        const float4 qv = *(const float4*)&qh_s[n * 16 + d0];
        float x0 = fmaxf(fmaf(qv.x - kv.x + pv.x, cav.x, cbv.x), 0.f);
        float x1 = fmaxf(fmaf(qv.y - kv.y + pv.y, cav.y, cbv.y), 0.f);
        float x2 = fmaxf(fmaf(qv.z - kv.z + pv.z, cav.z, cbv.z), 0.f);
        float x3 = fmaxf(fmaf(qv.w - kv.w + pv.w, cav.w, cbv.w), 0.f);
        float acc = x0 * w2v.x + x1 * w2v.y + x2 * w2v.z + x3 * w2v.w;
        acc += __shfl_xor(acc, 1, 32);
        acc += __shfl_xor(acc, 2, 32);
        if (lq == 0) s_s[n * N_SEQ + m] = acc + sbias;
    }
    __syncthreads();

    // ---- softmax over m: 2 rows per wave ----
    for (int r = wave * 2; r < wave * 2 + 2; ++r) {
        float mx = -3.0e38f;
        for (int i = lane; i < N_SEQ; i += 32) mx = fmaxf(mx, s_s[r * N_SEQ + i]);
        for (int o = 16; o > 0; o >>= 1) mx = fmaxf(mx, __shfl_xor(mx, o, 32));
        float sum = 0.f;
        for (int i = lane; i < N_SEQ; i += 32) {
            const float e = __expf(s_s[r * N_SEQ + i] - mx);
            s_s[r * N_SEQ + i] = e;
            sum += e;
        }
        for (int o = 16; o > 0; o >>= 1) sum += __shfl_xor(sum, o, 32);
        const float inv = 1.0f / sum;
        for (int i = lane; i < N_SEQ; i += 32)
            attn_s[r * N_SEQ + i] = (_Float16)(s_s[r * N_SEQ + i] * inv);
    }
    __syncthreads();

    // ---- attn[16,1024] @ V[1024,16] via WMMA; K split across 8 waves ----
    // Fragment halves are contiguous+16B aligned -> two ds_load_b128 each.
    const int row = lane & 15, hi = lane >> 4, kb = hi * 8;
    v8f accv = {};
#pragma unroll
    for (int kk = 0; kk < 4; ++kk) {
        const int k0 = (wave * 4 + kk) * 32;
        const _Float16* ap = &attn_s[row * N_SEQ + k0 + kb];   // A: row n = lane%16
        const _Float16* bp = &vhT   [row * N_SEQ + k0 + kb];   // B: col d = lane%16
        const v8h a0 = *(const v8h*)ap;
        const v8h a1 = *(const v8h*)(ap + 16);
        const v8h b0 = *(const v8h*)bp;
        const v8h b1 = *(const v8h*)(bp + 16);
        const v16h a  = __builtin_shufflevector(a0, a1, 0,1,2,3,4,5,6,7,8,9,10,11,12,13,14,15);
        const v16h bf = __builtin_shufflevector(b0, b1, 0,1,2,3,4,5,6,7,8,9,10,11,12,13,14,15);
        accv = __builtin_amdgcn_wmma_f32_16x16x32_f16(
            false, a, false, bf, (short)0, accv, false, false);
    }
#pragma unroll
    for (int j = 0; j < 8; ++j)
        atomicAdd(&y_tile[(j + hi * 8) * 16 + row], accv[j]);  // ds_add_f32
    __syncthreads();

    // y layout [B,N,H,D] so that reshape(B,N,C) has c = h*16+d
    {
        const int n = t >> 4, d = t & 15;
        y[(size_t)((b * N_SEQ + n0 + n) * 4 + h) * 16 + d] = y_tile[t];
    }
}

// ---------------------------------------------------------------------------
extern "C" void kernel_launch(void* const* d_in, const int* in_sizes, int n_in,
                              void* d_out, int out_size, void* d_ws, size_t ws_size,
                              hipStream_t stream)
{
    const float* q     = (const float*)d_in[0];
    const float* k     = (const float*)d_in[1];
    const float* v     = (const float*)d_in[2];
    const float* pos   = (const float*)d_in[3];
    const float* Wq    = (const float*)d_in[4];
    const float* Wk    = (const float*)d_in[5];
    const float* Wv    = (const float*)d_in[6];
    const float* w1    = (const float*)d_in[7];
    const float* b1    = (const float*)d_in[8];
    const float* bn1_g = (const float*)d_in[9];
    const float* bn1_b = (const float*)d_in[10];
    const float* bn1_m = (const float*)d_in[11];
    const float* bn1_v = (const float*)d_in[12];
    const float* w2    = (const float*)d_in[13];
    const float* b2    = (const float*)d_in[14];
    const float* bn2_g = (const float*)d_in[15];
    const float* bn2_b = (const float*)d_in[16];
    const float* bn2_m = (const float*)d_in[17];
    const float* bn2_v = (const float*)d_in[18];
    const float* Wp    = (const float*)d_in[19];
    const float* bp    = (const float*)d_in[20];

    float* ws = (float*)d_ws;                 // needs 4 * 2048*64 * 4B = 2 MB
    float* qh = ws;
    float* kh = ws + 131072;
    float* vp = ws + 262144;
    float* yb = ws + 393216;

    gemm64_wmma<<<64, 256, 0, stream>>>(q, Wq, nullptr, qh, 2048);
    gemm64_wmma<<<64, 256, 0, stream>>>(k, Wk, nullptr, kh, 2048);
    gemm64_wmma<<<64, 256, 0, stream>>>(v, Wv, nullptr, vp, 2048);

    attn_fused<<<512, 256, 0, stream>>>(qh, kh, vp, pos,
                                        w1, b1, bn1_g, bn1_b, bn1_m, bn1_v,
                                        w2, b2, bn2_g, bn2_b, bn2_m, bn2_v, yb);

    gemm64_wmma<<<64, 256, 0, stream>>>(yb, Wp, bp, (float*)d_out, 2048);
}
